// FeatExtractor_63015760167691
// MI455X (gfx1250) — matrix operand order repaired and checked by reading, not compiled, not run on hardware
//
#include <hip/hip_runtime.h>

typedef __attribute__((ext_vector_type(16))) _Float16 v16h;
typedef __attribute__((ext_vector_type(4)))  _Float16 v4h;
typedef __attribute__((ext_vector_type(8)))  float    v8f;
typedef __attribute__((ext_vector_type(4)))  unsigned int v4u;

#define B_   32
#define H_   128          // input H=W
#define C0   12           // 3*NUM_FRAMES
#define C0P  16           // padded input channels
#define C1   64           // conv1 out channels
#define HP   64           // pooled spatial (conv2 spatial)
#define C2   128          // conv2 out channels
#define O_   32           // NUM_OBJECTS

// workspace layout (halves)
#define IMGF_HALVES ((size_t)B_ * 130 * 130 * C0P)   //  8,652,800 (17.3 MB)
#define POOL_HALVES ((size_t)B_ * 66 * 66 * C1)      //  8,921,088 (17.8 MB)
#define A1_HALVES   ((size_t)4 * 5 * 32 * 16)        //  10,240
#define A2_HALVES   ((size_t)8 * 18 * 32 * 16)       //  73,728

// ---------------------------------------------------------------------------
__global__ void zero_out_kernel(float* __restrict__ out, int n) {
  int i = blockIdx.x * blockDim.x + threadIdx.x;
  if (i < n) out[i] = 0.0f;
}

__global__ void zero_ws_kernel(v4u* __restrict__ p, long nvec) {
  long i = (long)blockIdx.x * blockDim.x + threadIdx.x;
  if (i < nvec) p[i] = (v4u){0u, 0u, 0u, 0u};
}

// ---------------------------------------------------------------------------
// images (B,12,128,128) f32 NCHW -> f16 NHWC, /255, 16-ch padded, 1px halo.
// ---------------------------------------------------------------------------
__global__ void repack_images_kernel(const float* __restrict__ img,
                                     _Float16* __restrict__ imgf) {
  int i = blockIdx.x * 256 + threadIdx.x;     // 32*128*128 pixels
  int x = i & 127, y = (i >> 7) & 127, b = i >> 14;
  const float* src = img + (size_t)b * C0 * H_ * H_ + y * H_ + x;
  v16h v = {};
  #pragma unroll
  for (int c = 0; c < C0; ++c)
    v[c] = (_Float16)(src[(size_t)c * H_ * H_] * (1.0f / 255.0f));
  *(v16h*)(imgf + (((size_t)b * 130 + (y + 1)) * 130 + (x + 1)) * C0P) = v;
}

// ---------------------------------------------------------------------------
// conv1_w (64,12,3,3) -> WMMA A-fragments, k-order = tap*16 + inc.
// ---------------------------------------------------------------------------
__global__ void repack_w1_kernel(const float* __restrict__ w,
                                 _Float16* __restrict__ a1) {
  int i = blockIdx.x * 256 + threadIdx.x;     // 10240 elements
  if (i >= (int)A1_HALVES) return;
  int j = i & 15, lane = (i >> 4) & 31;
  int t = (i >> 9) % 5, cg = (i >> 9) / 5;
  int k = t * 32 + (lane >> 4) * 16 + j;
  int tap = k >> 4, inc = k & 15;
  int outc = cg * 16 + (lane & 15);
  float v = (tap < 9 && inc < C0) ? w[(size_t)outc * (C0 * 9) + inc * 9 + tap]
                                  : 0.0f;
  a1[i] = (_Float16)v;
}

// ---------------------------------------------------------------------------
// conv2_w (128,64,3,3) -> WMMA A-fragments, k-order = tap*64 + inc.
// ---------------------------------------------------------------------------
__global__ void repack_w2_kernel(const float* __restrict__ w,
                                 _Float16* __restrict__ a2) {
  int i = blockIdx.x * 256 + threadIdx.x;     // 73728 elements
  if (i >= (int)A2_HALVES) return;
  int j = i & 15, lane = (i >> 4) & 31;
  int t = (i >> 9) % 18, ct = (i >> 9) / 18;
  int k = t * 32 + (lane >> 4) * 16 + j;
  int tap = k >> 6, inc = k & 63;
  int outc = ct * 16 + (lane & 15);
  a2[i] = (_Float16)w[(size_t)outc * (C1 * 9) + inc * 9 + tap];
}

// ---------------------------------------------------------------------------
// Kernel A: conv1 (5 WMMA k-steps) + ReLU + 2x2 maxpool -> pooled NHWC f16
// (66x66 halo-padded). Block = 8 waves: cg(4 ch-groups) x px(2 col halves).
// ---------------------------------------------------------------------------
__global__ __launch_bounds__(256) void conv1_pool_kernel(
    const _Float16* __restrict__ imgf, const _Float16* __restrict__ a1,
    const float* __restrict__ bias, _Float16* __restrict__ pooled) {
  __shared__ float tile[C1 * 2 * 16];

  int blk = blockIdx.x;
  int pcg = blk & 7;
  int pr  = (blk >> 3) & 63;
  int b   = blk >> 9;

  int lane = threadIdx.x & 31;
  int wave = threadIdx.x >> 5;
  int cg   = wave & 3;
  int px   = wave >> 2;
  int mlo  = lane & 15;
  int hi   = lane >> 4;

  int n  = mlo;
  int cy = 2 * pr + (n >> 3);
  int cx = 16 * pcg + px * 8 + (n & 7);

  const _Float16* Af = a1 + ((size_t)(cg * 5) * 32 + lane) * 16;
  size_t imgb = (size_t)b * 130 * 130;

  v8f acc = {};
  #pragma unroll
  for (int t = 0; t < 5; ++t) {
    v16h A = *(const v16h*)(Af + (size_t)t * 32 * 16);
    int tap = 2 * t + hi;
    int kr = tap / 3, kc = tap - kr * 3;
    size_t off = (tap > 8) ? imgb
                           : imgb + (size_t)(cy + kr) * 130 + (cx + kc);
    v16h Bv = *(const v16h*)(imgf + off * C0P);
    acc = __builtin_amdgcn_wmma_f32_16x16x32_f16(
        false, A, false, Bv, (short)0, acc, false, false);
  }

  int row = n >> 3;
  int col = px * 8 + (n & 7);
  #pragma unroll
  for (int r = 0; r < 8; ++r) {
    int m = r + 8 * hi;
    int outc = cg * 16 + m;
    float v = acc[r] + bias[outc];
    tile[(outc * 2 + row) * 16 + col] = v > 0.0f ? v : 0.0f;
  }
  __syncthreads();

  if (threadIdx.x < 128) {
    int pc = threadIdx.x >> 4;
    int c0 = (threadIdx.x & 15) * 4;
    v4h o;
    #pragma unroll
    for (int q = 0; q < 4; ++q) {
      const float* tp = &tile[((c0 + q) * 2) * 16 + 2 * pc];
      o[q] = (_Float16)fmaxf(fmaxf(tp[0], tp[1]), fmaxf(tp[16], tp[17]));
    }
    *(v4h*)(pooled +
            (((size_t)b * 66 + (pr + 1)) * 66 + (8 * pcg + pc + 1)) * C1 + c0) = o;
  }
}

// ---------------------------------------------------------------------------
// Kernel B: conv2 (18 k-steps, K=576) with 2x2 register blocking: each wave
// computes (2 channel-tiles) x (2 pixel-tiles) = 4 WMMA tiles. Per k-step:
// 2 A-fragment loads + 2 B-fragment loads feed 4 v_wmma -> 32 B/WMMA
// (~16 FLOP/byte), solidly compute-bound. Block = 8 waves covers one full
// output row: 128 ch x 64 px. ReLU + fused ROI segment-max epilogue.
// ---------------------------------------------------------------------------
__global__ __launch_bounds__(256) void conv2_roipool_kernel(
    const _Float16* __restrict__ pooled, const _Float16* __restrict__ a2,
    const float* __restrict__ bias, const int* __restrict__ rois,
    float* __restrict__ out) {
  int blk = blockIdx.x;           // grid = B_*64: one output row per block
  int y   = blk & 63;
  int b   = blk >> 6;

  int lane = threadIdx.x & 31;
  int wave = threadIdx.x >> 5;
  int ctp  = wave & 3;            // channel-tile pair: ct = 2*ctp, 2*ctp+1
  int ptp  = wave >> 2;           // pixel-tile pair: x base 32*ptp
  int mlo  = lane & 15;
  int hi   = lane >> 4;

  int x0 = 32 * ptp + mlo;        // lane's pixel in tile 0 (tile 1 = +16)

  // warm the roi line for the epilogue (global_prefetch_b8)
  __builtin_prefetch(rois + (size_t)b * H_ * H_ + (2 * y) * H_ + 2 * x0, 0, 1);

  const _Float16* Af0 = a2 + ((size_t)((2 * ctp) * 18) * 32 + lane) * 16;
  const _Float16* Af1 = Af0 + (size_t)18 * 32 * 16;
  const _Float16* pb  = pooled + (size_t)b * 66 * 66 * C1;

  v8f acc00 = {}, acc01 = {}, acc10 = {}, acc11 = {};
  #pragma unroll
  for (int t = 0; t < 18; ++t) {
    const int tap = t >> 1;                 // compile-time after unroll
    const int kr = tap / 3, kc = tap % 3;   // compile-time
    v16h A0 = *(const v16h*)(Af0 + (size_t)t * 32 * 16);
    v16h A1 = *(const v16h*)(Af1 + (size_t)t * 32 * 16);
    const _Float16* bp =
        pb + ((size_t)(y + kr) * 66 + (x0 + kc)) * C1 + (t & 1) * 32 + hi * 16;
    v16h B0 = *(const v16h*)(bp);
    v16h B1 = *(const v16h*)(bp + (size_t)16 * C1);   // pixel tile +16 cols
    acc00 = __builtin_amdgcn_wmma_f32_16x16x32_f16(
        false, A0, false, B0, (short)0, acc00, false, false);
    acc01 = __builtin_amdgcn_wmma_f32_16x16x32_f16(
        false, A0, false, B1, (short)0, acc01, false, false);
    acc10 = __builtin_amdgcn_wmma_f32_16x16x32_f16(
        false, A1, false, B0, (short)0, acc10, false, false);
    acc11 = __builtin_amdgcn_wmma_f32_16x16x32_f16(
        false, A1, false, B1, (short)0, acc11, false, false);
  }

  // fused ROI segment-max (post-ReLU >= 0 vs zero-initialized out buffer)
  const int* rb = rois + (size_t)b * H_ * H_ + (2 * y) * H_;
  int id0 = rb[2 * x0];
  int id1 = rb[2 * (x0 + 16)];
  float* ob = out + (size_t)b * O_ * C2;
  #pragma unroll
  for (int ci = 0; ci < 2; ++ci) {
    v8f aP0 = ci ? acc10 : acc00;
    v8f aP1 = ci ? acc11 : acc01;
    int ctbase = (2 * ctp + ci) * 16;
    #pragma unroll
    for (int r = 0; r < 8; ++r) {
      int outc = ctbase + r + 8 * hi;
      float bz = bias[outc];
      float v0 = fmaxf(aP0[r] + bz, 0.0f);
      float v1 = fmaxf(aP1[r] + bz, 0.0f);
      atomicMax(ob + (size_t)id0 * C2 + outc, v0);
      atomicMax(ob + (size_t)id1 * C2 + outc, v1);
    }
  }
}

// ---------------------------------------------------------------------------
extern "C" void kernel_launch(void* const* d_in, const int* in_sizes, int n_in,
                              void* d_out, int out_size, void* d_ws, size_t ws_size,
                              hipStream_t stream) {
  const float* images = (const float*)d_in[0];
  const int*   rois   = (const int*)d_in[1];
  const float* w1     = (const float*)d_in[2];
  const float* b1     = (const float*)d_in[3];
  const float* w2     = (const float*)d_in[4];
  const float* b2     = (const float*)d_in[5];
  float* out = (float*)d_out;

  _Float16* imgf  = (_Float16*)d_ws;
  _Float16* pool  = imgf + IMGF_HALVES;
  _Float16* a1    = pool + POOL_HALVES;
  _Float16* a2    = a1 + A1_HALVES;

  long zvec = (long)(IMGF_HALVES + POOL_HALVES) / 8;   // 16B chunks
  zero_ws_kernel<<<(int)((zvec + 255) / 256), 256, 0, stream>>>((v4u*)d_ws, zvec);
  zero_out_kernel<<<(out_size + 255) / 256, 256, 0, stream>>>(out, out_size);

  repack_images_kernel<<<(B_ * H_ * H_) / 256, 256, 0, stream>>>(images, imgf);
  repack_w1_kernel<<<((int)A1_HALVES + 255) / 256, 256, 0, stream>>>(w1, a1);
  repack_w2_kernel<<<((int)A2_HALVES + 255) / 256, 256, 0, stream>>>(w2, a2);

  conv1_pool_kernel<<<B_ * 64 * 8, 256, 0, stream>>>(imgf, a1, b1, pool);
  conv2_roipool_kernel<<<B_ * 64, 256, 0, stream>>>(pool, a2, b2, rois, out);
}
